// SDW_39213051413015
// MI455X (gfx1250) — matrix operand order repaired and checked by reading, not compiled, not run on hardware
//
#include <hip/hip_runtime.h>
#include <cstdint>
#include <cstddef>

typedef __bf16 bf16;
typedef __attribute__((ext_vector_type(16))) __bf16 v16bf;
typedef __attribute__((ext_vector_type(8)))  float  v8f;

#define WMMA_BF16(A_, B_, C_) \
  __builtin_amdgcn_wmma_f32_16x16x32_bf16(false, (A_), false, (B_), (short)0, (C_), false, false)

static constexpr int BATCH = 32;
static constexpr int NTOK  = 256;   // N images per set
static constexpr int DIN   = 1024;
static constexpr int DM    = 512;   // model dim (= L)
static constexpr int NHEAD = 8;
static constexpr int DH    = 64;
static constexpr int NIND  = 16;
static constexpr float ATT_SCALE = 0.04419417382415922f; // 1/sqrt(512)

// ======================= device helpers =======================
__device__ __forceinline__ float wave_sum(float v) {
  #pragma unroll
  for (int o = 16; o > 0; o >>= 1) v += __shfl_xor(v, o, 32);
  return v;
}
__device__ __forceinline__ float wave_max(float v) {
  #pragma unroll
  for (int o = 16; o > 0; o >>= 1) v = fmaxf(v, __shfl_xor(v, o, 32));
  return v;
}

// ======================= elementwise kernels =======================
// convert f32 W[K,N] -> bf16 W^T[N,K] (contiguous reads, strided writes)
__global__ void k_cvt_t(const float* __restrict__ s, bf16* __restrict__ d, int K, int N) {
  size_t i = (size_t)blockIdx.x * blockDim.x + threadIdx.x;
  size_t tot = (size_t)K * N;
  if (i >= tot) return;
  int k = (int)(i / N), n = (int)(i % N);
  d[(size_t)n * K + k] = (bf16)s[i];
}
__global__ void k_bcast(const float* __restrict__ s, bf16* __restrict__ d,
                        int srows, int drows, int cols) {
  size_t i = (size_t)blockIdx.x * blockDim.x + threadIdx.x;
  size_t tot = (size_t)drows * cols;
  if (i < tot) {
    int r = (int)(i / cols), c = (int)(i % cols);
    d[i] = (bf16)s[(size_t)(r % srows) * cols + c];
  }
}
__global__ void k_zero_f32(float* p, size_t n) {
  size_t i = (size_t)blockIdx.x * blockDim.x + threadIdx.x;
  if (i < n) p[i] = 0.f;
}
__global__ void k_zero_bf16(bf16* p, size_t n) {
  size_t i = (size_t)blockIdx.x * blockDim.x + threadIdx.x;
  if (i < n) p[i] = (bf16)0.f;
}
// concat(img, broadcast(txt)) -> bf16 (8192 x 2048)
__global__ void k_concat(const float* __restrict__ img, const float* __restrict__ txt,
                         bf16* __restrict__ out) {
  size_t i = (size_t)blockIdx.x * blockDim.x + threadIdx.x;
  size_t tot = (size_t)BATCH * NTOK * 2 * DIN;
  if (i >= tot) return;
  size_t row = i >> 11;           // /2048
  int c = (int)(i & 2047);
  int b = (int)(row >> 8);        // /256
  float v = (c < DIN) ? img[row * DIN + c] : txt[(size_t)b * DIN + (c - DIN)];
  out[i] = (bf16)v;
}
// per-(batch,head) transpose of V: vT[z][n][k] = v[b*Nk+k][h*64+n], k contiguous
__global__ void k_vt(const bf16* __restrict__ vbuf, bf16* __restrict__ vT,
                     int Nk, int NkPad) {
  size_t i = (size_t)blockIdx.x * blockDim.x + threadIdx.x;
  size_t tot = (size_t)BATCH * NHEAD * DH * NkPad;
  if (i >= tot) return;
  int k = (int)(i % NkPad);
  int n = (int)((i / NkPad) % DH);
  int z = (int)(i / ((size_t)NkPad * DH));
  int b = z / NHEAD, h = z % NHEAD;
  vT[i] = vbuf[(size_t)(b * Nk + k) * DM + h * DH + n];
}

// ======================= WMMA GEMM =======================
// C[M,N] = act(A[M,K] @ W + bias) (+ residual R), W given TRANSPOSED: Wt[N,K].
// A, Wt bf16 row-major (ldw = K for Wt).  Epilogue: bias -> relu -> +R.
// Block: 128 threads (4 waves). Tile BM=64 (16 rows/wave), BN=64, BK=32.
// Staging is software-pipelined: next K-tile is fetched into registers while
// the current LDS tile feeds the WMMAs.
__global__ __launch_bounds__(128)
void k_gemm(const bf16* __restrict__ A, int lda,
            const bf16* __restrict__ Wt, int ldw,
            const float* __restrict__ bias,
            const float* __restrict__ R, int ldr,
            float* __restrict__ Cf, bf16* __restrict__ Cb, int ldc,
            int K, int relu) {
  __shared__ bf16 As[64 * 32];
  __shared__ bf16 Bs[64 * 32];   // Bs[n][k]
  const int tid  = threadIdx.x;
  const int wave = tid >> 5, lane = tid & 31;
  const int r = lane & 15, hi = lane >> 4;
  const int m0 = blockIdx.y * 64, n0 = blockIdx.x * 64;

  const int arow = tid >> 1, acb = (tid & 1) * 16;   // A: 64 rows x 32 cols
  const int bn   = tid & 63, bkh = tid >> 6;         // B: 64 n-rows x 32 k-cols

  v8f acc[4];
  #pragma unroll
  for (int nt = 0; nt < 4; ++nt) acc[nt] = (v8f){0.f,0.f,0.f,0.f,0.f,0.f,0.f,0.f};

  v16bf areg, breg;
  { // preload first tile into registers
    const bf16* sa = A  + (size_t)(m0 + arow) * lda + acb;
    const bf16* sb = Wt + (size_t)(n0 + bn)  * ldw + bkh * 16;
    #pragma unroll
    for (int i = 0; i < 16; ++i) { areg[i] = sa[i]; breg[i] = sb[i]; }
  }

  for (int k0 = 0; k0 < K; k0 += 32) {
    { // commit staged registers to LDS (vectorized contiguous stores)
      bf16* da = As + arow * 32 + acb;
      bf16* db = Bs + bn * 32 + bkh * 16;
      #pragma unroll
      for (int i = 0; i < 16; ++i) { da[i] = areg[i]; db[i] = breg[i]; }
    }
    __syncthreads();
    if (k0 + 32 < K) { // fetch next tile; latency hidden behind WMMA block
      const bf16* sa = A  + (size_t)(m0 + arow) * lda + k0 + 32 + acb;
      const bf16* sb = Wt + (size_t)(n0 + bn)  * ldw + k0 + 32 + bkh * 16;
      #pragma unroll
      for (int i = 0; i < 16; ++i) { areg[i] = sa[i]; breg[i] = sb[i]; }
      if (k0 + 64 < K) {
        __builtin_prefetch(A  + (size_t)(m0 + arow) * lda + k0 + 64, 0, 3);
        __builtin_prefetch(Wt + (size_t)(n0 + bn)  * ldw + k0 + 64, 0, 3);
      }
    }
    // A fragment: contiguous k = 16*hi + i (matches B)
    v16bf a;
    const bf16* ar = As + (wave * 16 + r) * 32;
    #pragma unroll
    for (int i = 0; i < 16; ++i) a[i] = ar[hi * 16 + i];
    #pragma unroll
    for (int nt = 0; nt < 4; ++nt) {
      v16bf bfr;
      const bf16* br = Bs + (nt * 16 + r) * 32 + hi * 16;  // contiguous per lane
      #pragma unroll
      for (int i = 0; i < 16; ++i) bfr[i] = br[i];
      acc[nt] = WMMA_BF16(a, bfr, acc[nt]);
    }
    __syncthreads();
  }
  // epilogue
  #pragma unroll
  for (int nt = 0; nt < 4; ++nt) {
    #pragma unroll
    for (int j = 0; j < 8; ++j) {
      int m = m0 + wave * 16 + j + 8 * hi;
      int n = n0 + nt * 16 + r;
      float v = acc[nt][j];
      if (bias) v += bias[n];
      if (relu) v = fmaxf(v, 0.f);
      if (R)    v += R[(size_t)m * ldr + n];
      if (Cf) Cf[(size_t)m * ldc + n] = v;
      if (Cb) Cb[(size_t)m * ldc + n] = (bf16)v;
    }
  }
}

// ======================= attention kernels =======================
// scores[z][i][j] = scale * q[b*Nq+i, h*64: ] . k[b*Nk+j, h*64: ]   (K = 64)
// grid: (NkPad/16, NqPad/16, B*H), block = 32 (one wave)
__global__ __launch_bounds__(32)
void k_scores(const bf16* __restrict__ q, const bf16* __restrict__ kbuf,
              float* __restrict__ scores, int Nq, int Nk, int NqPad, int NkPad,
              float scale) {
  const int lane = threadIdx.x, r = lane & 15, hi = lane >> 4;
  const int jt = blockIdx.x, it = blockIdx.y, z = blockIdx.z;
  const int b = z / NHEAD, h = z % NHEAD;
  v8f acc = (v8f){0.f,0.f,0.f,0.f,0.f,0.f,0.f,0.f};
  #pragma unroll
  for (int k0 = 0; k0 < DH; k0 += 32) {
    v16bf a, bfr;
    const bf16* qr = q + (size_t)(b * Nq + it * 16 + r) * DM + h * DH + k0;
    #pragma unroll
    for (int i = 0; i < 16; ++i) a[i] = qr[hi * 16 + i];
    const bf16* kr = kbuf + (size_t)(b * Nk + jt * 16 + r) * DM + h * DH + k0 + hi * 16;
    #pragma unroll
    for (int i = 0; i < 16; ++i) bfr[i] = kr[i];
    acc = WMMA_BF16(a, bfr, acc);
  }
  float* S = scores + (size_t)z * NqPad * NkPad;
  #pragma unroll
  for (int j = 0; j < 8; ++j)
    S[(size_t)(it * 16 + j + 8 * hi) * NkPad + jt * 16 + r] = acc[j] * scale;
}

// rowwise softmax over Nk, writes bf16 probs, zero-pads cols [Nk, NkPad)
__global__ __launch_bounds__(256)
void k_softmax(const float* __restrict__ S, bf16* __restrict__ P,
               int Nk, int NkPad, int rows) {
  int row  = (int)(((size_t)blockIdx.x * blockDim.x + threadIdx.x) >> 5);
  int lane = threadIdx.x & 31;
  if (row >= rows) return;
  const float* x = S + (size_t)row * NkPad;
  bf16* y = P + (size_t)row * NkPad;
  float m = -3.4e38f;
  for (int c = lane; c < Nk; c += 32) m = fmaxf(m, x[c]);
  m = wave_max(m);
  float s = 0.f;
  for (int c = lane; c < Nk; c += 32) s += __expf(x[c] - m);
  s = wave_sum(s);
  float inv = 1.f / s;
  for (int c = lane; c < NkPad; c += 32)
    y[c] = (c < Nk) ? (bf16)(__expf(x[c] - m) * inv) : (bf16)0.f;
}

// O[b*Nq+i, h*64+n] = q_res + sum_j probs[z][i][j] * vT[z][n][j]
// grid: (NqPad/16, B*H), block 32. Stores guarded by i < Nq.
__global__ __launch_bounds__(32)
void k_av(const bf16* __restrict__ probs, int NqPad, int NkPad,
          const bf16* __restrict__ vT, const float* __restrict__ qres,
          float* __restrict__ O, int Nq) {
  const int lane = threadIdx.x, r = lane & 15, hi = lane >> 4;
  const int mt = blockIdx.x, z = blockIdx.y;
  const int b = z / NHEAD, h = z % NHEAD;
  const bf16* Ap = probs + (size_t)z * NqPad * NkPad + (size_t)mt * 16 * NkPad;
  const bf16* Vz = vT + (size_t)z * DH * NkPad;
  v8f acc[4];
  #pragma unroll
  for (int nt = 0; nt < 4; ++nt) acc[nt] = (v8f){0.f,0.f,0.f,0.f,0.f,0.f,0.f,0.f};
  for (int kt = 0; kt < NkPad; kt += 32) {
    v16bf a;
    const bf16* ar = Ap + (size_t)r * NkPad + kt;
    #pragma unroll
    for (int i = 0; i < 16; ++i) a[i] = ar[hi * 16 + i];
    #pragma unroll
    for (int nt = 0; nt < 4; ++nt) {
      v16bf bfr;
      const bf16* vr = Vz + (size_t)(nt * 16 + r) * NkPad + kt + hi * 16;  // contiguous
      #pragma unroll
      for (int i = 0; i < 16; ++i) bfr[i] = vr[i];
      acc[nt] = WMMA_BF16(a, bfr, acc[nt]);
    }
  }
  #pragma unroll
  for (int nt = 0; nt < 4; ++nt) {
    #pragma unroll
    for (int j = 0; j < 8; ++j) {
      int m = mt * 16 + j + 8 * hi;
      if (m < Nq) {
        size_t row = (size_t)b * Nq + m;
        int col = h * DH + nt * 16 + r;
        O[row * DM + col] = acc[nt][j] + qres[row * DM + col];
      }
    }
  }
}

// LayerNorm over 512 cols; wave per row; writes f32 and/or bf16
__global__ __launch_bounds__(256)
void k_ln(const float* __restrict__ in, float* __restrict__ outf,
          bf16* __restrict__ outb, const float* __restrict__ g,
          const float* __restrict__ bt, int rows) {
  int row  = (int)(((size_t)blockIdx.x * blockDim.x + threadIdx.x) >> 5);
  int lane = threadIdx.x & 31;
  if (row >= rows) return;
  const float* x = in + (size_t)row * DM;
  float vals[16];
  float s = 0.f;
  #pragma unroll
  for (int i = 0; i < 16; ++i) { vals[i] = x[lane + i * 32]; s += vals[i]; }
  float mu = wave_sum(s) * (1.f / DM);
  float vs = 0.f;
  #pragma unroll
  for (int i = 0; i < 16; ++i) { float d = vals[i] - mu; vs += d * d; }
  float rinv = rsqrtf(wave_sum(vs) * (1.f / DM) + 1e-5f);
  #pragma unroll
  for (int i = 0; i < 16; ++i) {
    int c = lane + i * 32;
    float y = (vals[i] - mu) * rinv * g[c] + bt[c];
    if (outf) outf[(size_t)row * DM + c] = y;
    if (outb) outb[(size_t)row * DM + c] = (bf16)y;
  }
}

// out[b*256+n] = dot(Fx[row,:], wl[b,:])
__global__ __launch_bounds__(256)
void k_dot(const float* __restrict__ Fx, const float* __restrict__ wl,
           float* __restrict__ out) {
  int row  = (int)(((size_t)blockIdx.x * blockDim.x + threadIdx.x) >> 5);
  int lane = threadIdx.x & 31;
  if (row >= BATCH * NTOK) return;
  int b = row >> 8;
  float s = 0.f;
  for (int c = lane; c < DM; c += 32)
    s += Fx[(size_t)row * DM + c] * wl[(size_t)b * DM + c];
  s = wave_sum(s);
  if (lane == 0) out[row] = s;
}

// ======================= host side =======================
struct Lin  { const float *W, *b; };
struct MabP { Lin q, k, v, o; const float *g0, *b0, *g1, *b1; };
struct MabW { bf16 *q, *k, *v, *o; };

static void fill_mab_ins(MabP& m, void* const* in, int& c) {
  m.q.W=(const float*)in[c++]; m.q.b=(const float*)in[c++];
  m.k.W=(const float*)in[c++]; m.k.b=(const float*)in[c++];
  m.v.W=(const float*)in[c++]; m.v.b=(const float*)in[c++];
  m.o.W=(const float*)in[c++]; m.o.b=(const float*)in[c++];
  m.g0=(const float*)in[c++]; m.b0=(const float*)in[c++];
  m.g1=(const float*)in[c++]; m.b1=(const float*)in[c++];
}
static void fill_mab_sorted(MabP& m, void* const* in, int& c) {
  m.b0=(const float*)in[c++]; m.b1=(const float*)in[c++];
  m.g0=(const float*)in[c++]; m.g1=(const float*)in[c++];
  m.k.W=(const float*)in[c++]; m.k.b=(const float*)in[c++];
  m.o.W=(const float*)in[c++]; m.o.b=(const float*)in[c++];
  m.q.W=(const float*)in[c++]; m.q.b=(const float*)in[c++];
  m.v.W=(const float*)in[c++]; m.v.b=(const float*)in[c++];
}

static void gemm_launch(hipStream_t st, const bf16* A, int lda, const bf16* Wt, int ldw,
                        const float* bias, const float* R, int ldr,
                        float* Cf, bf16* Cb, int ldc, int M, int N, int K, int relu) {
  dim3 g((unsigned)(N / 64), (unsigned)(M / 64));
  k_gemm<<<g, 128, 0, st>>>(A, lda, Wt, ldw, bias, R, ldr, Cf, Cb, ldc, K, relu);
}

struct Scratch {
  float *qf, *scores, *Of;
  bf16 *qbf, *kbf, *vbf, *vT, *probs, *Obf;
};

static void run_mab(hipStream_t st, const MabP& p, const MabW& w, const Scratch& sc,
                    const bf16* Qin, int Mq, int Nq,
                    const bf16* Kin, int Mk, int Nk,
                    bf16* OutBf) {
  const int NqPad = (Nq + 15) & ~15;
  const int NkPad = (Nk + 31) & ~31;
  // linears (weights transposed; ldw = K = 512)
  gemm_launch(st, Qin, DM, w.q, DM, p.q.b, nullptr, 0, sc.qf, sc.qbf, DM, Mq, DM, DM, 0);
  gemm_launch(st, Kin, DM, w.k, DM, p.k.b, nullptr, 0, nullptr, sc.kbf, DM, Mk, DM, DM, 0);
  gemm_launch(st, Kin, DM, w.v, DM, p.v.b, nullptr, 0, nullptr, sc.vbf, DM, Mk, DM, DM, 0);
  // zero 64 tail rows (K-padding safety: 0 * finite = 0)
  size_t tail = (size_t)64 * DM;
  k_zero_bf16<<<(unsigned)((tail + 255) / 256), 256, 0, st>>>(sc.kbf + (size_t)Mk * DM, tail);
  k_zero_bf16<<<(unsigned)((tail + 255) / 256), 256, 0, st>>>(sc.vbf + (size_t)Mk * DM, tail);
  if (Mq == 64)  // ragged Nq=1 case: k_av only writes i < Nq
    k_zero_f32<<<(unsigned)((tail + 255) / 256), 256, 0, st>>>(sc.Of, tail);
  // per-head V transpose so A*V fragments are contiguous
  { size_t n = (size_t)BATCH * NHEAD * DH * NkPad;
    k_vt<<<(unsigned)((n + 255) / 256), 256, 0, st>>>(sc.vbf, sc.vT, Nk, NkPad); }
  // attention
  dim3 gs((unsigned)(NkPad / 16), (unsigned)(NqPad / 16), (unsigned)(BATCH * NHEAD));
  k_scores<<<gs, 32, 0, st>>>(sc.qbf, sc.kbf, sc.scores, Nq, Nk, NqPad, NkPad, ATT_SCALE);
  int rows = BATCH * NHEAD * NqPad;
  k_softmax<<<(unsigned)((rows + 7) / 8), 256, 0, st>>>(sc.scores, sc.probs, Nk, NkPad, rows);
  k_av<<<dim3((unsigned)(NqPad / 16), (unsigned)(BATCH * NHEAD)), 32, 0, st>>>(
      sc.probs, NqPad, NkPad, sc.vT, sc.qf, sc.Of, Nq);
  // LN0, o-linear (relu + residual), LN1
  k_ln<<<(unsigned)((Mq + 7) / 8), 256, 0, st>>>(sc.Of, sc.Of, sc.Obf, p.g0, p.b0, Mq);
  gemm_launch(st, sc.Obf, DM, w.o, DM, p.o.b, sc.Of, DM, sc.qf, nullptr, DM, Mq, DM, DM, 1);
  k_ln<<<(unsigned)((Mq + 7) / 8), 256, 0, st>>>(sc.qf, nullptr, OutBf, p.g1, p.b1, Mq);
}

extern "C" void kernel_launch(void* const* d_in, const int* in_sizes, int n_in,
                              void* d_out, int out_size, void* d_ws, size_t ws_size,
                              hipStream_t stream) {
  (void)out_size; (void)ws_size;
  // ---------- input mapping (insertion vs jax sorted-key order) ----------
  const float *img, *txt, *Sseed, *Iarr[4];
  Lin F[4], outL;
  MabP pma, sab[4], m0[4], m1[4];
  bool sorted = (n_in > 1 && in_sizes[1] == BATCH);  // num_images second -> sorted
  int c;
  if (!sorted) {
    img = (const float*)d_in[0]; txt = (const float*)d_in[1];
    c = 4;  // skip num_images, scale
    for (int i = 0; i < 4; ++i) { F[i].W=(const float*)d_in[c++]; F[i].b=(const float*)d_in[c++]; }
    for (int l = 0; l < 4; ++l) {
      Iarr[l] = (const float*)d_in[c++];
      fill_mab_ins(m0[l], d_in, c);
      fill_mab_ins(m1[l], d_in, c);
    }
    Sseed = (const float*)d_in[c++];
    fill_mab_ins(pma, d_in, c);
    for (int s = 0; s < 4; ++s) fill_mab_ins(sab[s], d_in, c);
    outL.W = (const float*)d_in[c++]; outL.b = (const float*)d_in[c++];
  } else {
    img = (const float*)d_in[0];
    c = 2;
    for (int i = 0; i < 4; ++i) { F[i].W=(const float*)d_in[c++]; F[i].b=(const float*)d_in[c++]; }
    Sseed = (const float*)d_in[c++];
    outL.W = (const float*)d_in[c++]; outL.b = (const float*)d_in[c++];
    fill_mab_sorted(pma, d_in, c);
    for (int s = 0; s < 4; ++s) fill_mab_sorted(sab[s], d_in, c);
    for (int l = 0; l < 4; ++l) {
      Iarr[l] = (const float*)d_in[c++];
      fill_mab_sorted(m0[l], d_in, c);
      fill_mab_sorted(m1[l], d_in, c);
    }
    txt = (const float*)d_in[n_in - 1];
  }

  // ---------- workspace layout ----------
  char* base = (char*)d_ws;
  size_t off = 0;
  auto alloc = [&](size_t bytes) -> void* {
    off = (off + 255) & ~(size_t)255;
    void* p = base + off; off += bytes; return p;
  };
  const size_t W2 = (size_t)DM * DM;              // 512*512
  bf16 *wF[4];
  wF[0] = (bf16*)alloc((size_t)2 * DIN * 1024 * 2);
  wF[1] = (bf16*)alloc((size_t)1024 * 1024 * 2);
  wF[2] = (bf16*)alloc((size_t)1024 * 512 * 2);
  wF[3] = (bf16*)alloc(W2 * 2);
  MabW w0[4], w1[4], wPma, wSab[4];
  auto allocMabW = [&](MabW& w) {
    w.q=(bf16*)alloc(W2*2); w.k=(bf16*)alloc(W2*2); w.v=(bf16*)alloc(W2*2); w.o=(bf16*)alloc(W2*2);
  };
  for (int l = 0; l < 4; ++l) { allocMabW(w0[l]); allocMabW(w1[l]); }
  allocMabW(wPma);
  for (int s = 0; s < 4; ++s) allocMabW(wSab[s]);
  bf16* wOut = (bf16*)alloc(W2 * 2);
  bf16* Ib[4];
  for (int l = 0; l < 4; ++l) Ib[l] = (bf16*)alloc((size_t)BATCH * NIND * DM * 2);
  bf16* Sb = (bf16*)alloc((size_t)64 * DM * 2);

  const size_t MROW = (size_t)BATCH * NTOK;       // 8192
  // overlapped region: MLP view vs attention-scratch view
  off = (off + 255) & ~(size_t)255;
  size_t reg = off;
  bf16* xcat = (bf16*)(base + reg);
  bf16* hA   = (bf16*)((char*)xcat + MROW * 2048 * 2);
  bf16* hB   = (bf16*)((char*)hA + MROW * 1024 * 2);
  size_t mlpBytes = MROW * 2048 * 2 + 2 * MROW * 1024 * 2;
  char* pA = base + reg;
  Scratch sc;
  sc.qf  = (float*)pA; pA += MROW * DM * 4;
  sc.qbf = (bf16*)pA;  pA += MROW * DM * 2;
  sc.kbf = (bf16*)pA;  pA += (MROW + 64) * DM * 2;
  sc.vbf = (bf16*)pA;  pA += (MROW + 64) * DM * 2;
  sc.vT  = (bf16*)pA;  pA += (size_t)BATCH * NHEAD * DH * 256 * 2;  // max NkPad = 256
  sc.Of  = (float*)pA; pA += MROW * DM * 4;
  sc.Obf = (bf16*)pA;  pA += MROW * DM * 2;
  size_t attBytes = (size_t)(pA - (base + reg));
  off = reg + (mlpBytes > attBytes ? mlpBytes : attBytes);

  float* Fx   = (float*)alloc(MROW * DM * 4);
  bf16*  Xbf0 = (bf16*)alloc(MROW * DM * 2);
  bf16*  Xbf1 = (bf16*)alloc(MROW * DM * 2);
  sc.scores = (float*)alloc((size_t)2097152 * 4);  // max B*H*NqPad*NkPad
  sc.probs  = (bf16*)alloc((size_t)2097152 * 2);
  bf16* Hmbf = (bf16*)alloc((size_t)BATCH * NIND * DM * 2);
  bf16* Ybf0 = (bf16*)alloc((size_t)64 * DM * 2);
  bf16* Ybf1 = (bf16*)alloc((size_t)64 * DM * 2);
  float* wlf = (float*)alloc((size_t)64 * DM * 4);

  // ---------- weight conversion (f32 [K,N] -> bf16 transposed [N,K]) ----------
  auto cvtT = [&](const float* s, bf16* d, int K, int N) {
    size_t n = (size_t)K * N;
    k_cvt_t<<<(unsigned)((n + 255) / 256), 256, 0, stream>>>(s, d, K, N);
  };
  cvtT(F[0].W, wF[0], 2048, 1024);
  cvtT(F[1].W, wF[1], 1024, 1024);
  cvtT(F[2].W, wF[2], 1024, 512);
  cvtT(F[3].W, wF[3], 512, 512);
  auto cvtMab = [&](const MabP& p, const MabW& w) {
    cvtT(p.q.W, w.q, 512, 512); cvtT(p.k.W, w.k, 512, 512);
    cvtT(p.v.W, w.v, 512, 512); cvtT(p.o.W, w.o, 512, 512);
  };
  for (int l = 0; l < 4; ++l) { cvtMab(m0[l], w0[l]); cvtMab(m1[l], w1[l]); }
  cvtMab(pma, wPma);
  for (int s = 0; s < 4; ++s) cvtMab(sab[s], wSab[s]);
  cvtT(outL.W, wOut, 512, 512);
  for (int l = 0; l < 4; ++l) {
    size_t n = (size_t)BATCH * NIND * DM;
    k_bcast<<<(unsigned)((n + 255) / 256), 256, 0, stream>>>(Iarr[l], Ib[l], NIND, BATCH * NIND, DM);
  }
  { size_t n = (size_t)64 * DM;
    k_bcast<<<(unsigned)((n + 255) / 256), 256, 0, stream>>>(Sseed, Sb, 1, 64, DM); }

  // ---------- F MLP (ldw = K of each layer, weights transposed) ----------
  { size_t n = MROW * 2048;
    k_concat<<<(unsigned)((n + 255) / 256), 256, 0, stream>>>(img, txt, xcat); }
  gemm_launch(stream, xcat, 2048, wF[0], 2048, F[0].b, nullptr, 0, nullptr, hA, 1024,
              (int)MROW, 1024, 2048, 1);
  gemm_launch(stream, hA, 1024, wF[1], 1024, F[1].b, nullptr, 0, nullptr, hB, 1024,
              (int)MROW, 1024, 1024, 1);
  gemm_launch(stream, hB, 1024, wF[2], 1024, F[2].b, nullptr, 0, nullptr, hA, 512,
              (int)MROW, 512, 1024, 1);
  gemm_launch(stream, hA, 512, wF[3], 512, F[3].b, nullptr, 0, Fx, Xbf0, 512,
              (int)MROW, 512, 512, 0);

  // ---------- ISAB encoder ----------
  bf16 *Xcur = Xbf0, *Xnxt = Xbf1;
  for (int l = 0; l < 4; ++l) {
    run_mab(stream, m0[l], w0[l], sc, Ib[l], BATCH * NIND, NIND, Xcur, (int)MROW, NTOK, Hmbf);
    run_mab(stream, m1[l], w1[l], sc, Xcur, (int)MROW, NTOK, Hmbf, BATCH * NIND, NIND, Xnxt);
    bf16* t = Xcur; Xcur = Xnxt; Xnxt = t;
  }

  // ---------- PMA + SAB decoder ----------
  run_mab(stream, pma, wPma, sc, Sb, 64, 1, Xcur, (int)MROW, NTOK, Ybf0);
  bf16 *Ycur = Ybf0, *Ynxt = Ybf1;
  for (int s = 0; s < 4; ++s) {
    run_mab(stream, sab[s], wSab[s], sc, Ycur, 64, 1, Ycur, 64, 1, Ynxt);
    bf16* t = Ycur; Ycur = Ynxt; Ynxt = t;
  }
  gemm_launch(stream, Ycur, DM, wOut, DM, outL.b, nullptr, 0, wlf, nullptr, DM, 64, DM, DM, 0);

  // ---------- final weighted reduction ----------
  { unsigned g = (unsigned)((MROW * 32 + 255) / 256);
    k_dot<<<g, 256, 0, stream>>>(Fx, wlf, (float*)d_out); }
}
